// ATTN_IRT_19164144075377
// MI455X (gfx1250) — matrix operand order repaired and checked by reading, not compiled, not run on hardware
//
#include <hip/hip_runtime.h>
#include <hip/hip_bf16.h>
#include <math.h>

// ---------------------------------------------------------------------------
// Model constants
// ---------------------------------------------------------------------------
#define BB   16
#define TT   512
#define HH   1024
#define NH   16
#define HD   64
#define MROWS (BB*TT)          // 8192
#define NSK  50
#define NIT  1000
#define PAD_NEG (-4294967296.0f)
#define NEG_INF (-INFINITY)

typedef __attribute__((ext_vector_type(16))) _Float16 v16h;
typedef __attribute__((ext_vector_type(8)))  float    v8f;
typedef unsigned u32;
typedef __attribute__((ext_vector_type(4))) u32 su4;
typedef __attribute__((ext_vector_type(8))) u32 su8;

__device__ __forceinline__ v8f wmma_f16(v16h a, v16h b, v8f c) {
    // D = A(16x32 f16) x B(32x16 f16) + C(16x16 f32)
    return __builtin_amdgcn_wmma_f32_16x16x32_f16(
        /*neg_a=*/false, a, /*neg_b=*/false, b,
        /*c_mod=*/(short)0, c, /*reuse_a=*/false, /*reuse_b=*/false);
}

union FragU { uint4 u[2]; v16h v; };
__device__ __forceinline__ v16h make_frag(const void* p0, const void* p1) {
    FragU f;
    f.u[0] = *(const uint4*)p0;   // halfs K =  koff .. koff+7
    f.u[1] = *(const uint4*)p1;   // halfs K = 16+koff .. 16+koff+7
    return f.v;
}

// CDNA5 async global->LDS copy (16B per lane), tracked by ASYNCcnt.
__device__ __forceinline__ void async_load_b128(unsigned lds_addr,
                                                const _Float16* gaddr) {
    asm volatile("global_load_async_to_lds_b128 %0, %1, off"
                 :: "v"(lds_addr), "v"(gaddr)
                 : "memory");
}
__device__ __forceinline__ void wait_async0() {
    asm volatile("s_wait_asynccnt 0x0" ::: "memory");
}
__device__ __forceinline__ void wait_ds0() {
    asm volatile("s_wait_dscnt 0x0" ::: "memory");
}

// CDNA5 Tensor Data Mover: DMA a 2D tile (tile_w x tile_h elements of 2 bytes,
// row stride row_len elements) from global memory into LDS, with optional LDS
// destination padding from the D# pad fields. 2-group descriptor (<=2D).
// Tracked by TENSORcnt.
__device__ __forceinline__ void tdm_load_2d(u32 lds_addr, const void* gaddr,
                                            u32 tile_w, u32 tile_h,
                                            u32 row_len, u32 pad_cfg) {
    const uint64_t ga = (uint64_t)(size_t)gaddr;
    su4 g0;
    g0[0] = 1u;                                  // count=1 (valid), user mode
    g0[1] = lds_addr;                            // lds_addr (bytes)
    g0[2] = (u32)ga;                             // global_addr[31:0]
    g0[3] = ((u32)(ga >> 32) & 0x01FFFFFFu)      // global_addr[56:32]
          | 0x80000000u;                         // type=2 (bits 127:126 = 10b)
    su8 g1;
    g1[0] = (1u << 16) | pad_cfg;                // data_size=1 (2B) + pad cfg
    g1[1] = (row_len & 0xFFFFu) << 16;           // tensor_dim0[15:0]
    g1[2] = (row_len >> 16) & 0xFFFFu;           // tensor_dim0[31:16]; dim1 lo16=0
    g1[3] = 0x0010u                              // tensor_dim1[31:16] -> 1M rows
          | (tile_w << 16);                      // tile_dim0
    g1[4] = tile_h;                              // tile_dim1 (tile_dim2 = 0)
    g1[5] = row_len;                             // tensor_dim0_stride[31:0]
    g1[6] = 0u;                                  // stride hi / dim1_stride lo
    g1[7] = 0u;
    asm volatile("tensor_load_to_lds %0, %1" :: "s"(g0), "s"(g1) : "memory");
}
// pad: after every 16 DWORDs (one 32-half row) insert 4 DWORDs (8 halfs)
// -> LDS row stride of 40 halfs.
#define TDM_PAD_ROW32 ((1u << 20) | (3u << 22) | (3u << 25))

// ---------------------------------------------------------------------------
// f32 -> f16 conversion (weights)
// ---------------------------------------------------------------------------
__global__ void cvt_f16_kernel(const float* __restrict__ src,
                               _Float16* __restrict__ dst, int n) {
    int i = blockIdx.x * blockDim.x + threadIdx.x;
    int stride = gridDim.x * blockDim.x;
    for (; i < n; i += stride) dst[i] = (_Float16)src[i];
}

// ---------------------------------------------------------------------------
// item-item compatibility mask:  mask[i][j] = (Q^T Q)[i][j] > 0 ? 0 : -inf
// ---------------------------------------------------------------------------
__global__ void qtq_mask_kernel(const float* __restrict__ qmat,
                                float* __restrict__ mask) {
    int i = blockIdx.x;                       // 0..999
    for (int j = threadIdx.x; j < NIT; j += blockDim.x) {
        float d = 0.f;
        for (int s = 0; s < NSK; ++s)
            d += qmat[s * NIT + i] * qmat[s * NIT + j];
        mask[(size_t)i * NIT + j] = (d > 0.f) ? 0.f : NEG_INF;
    }
}

// ---------------------------------------------------------------------------
// embedding + sinusoidal position encoding -> x_f32 (residual) + x_f16
// ---------------------------------------------------------------------------
__global__ void embed_kernel(const int* __restrict__ prob_x,
                             const float* __restrict__ emb,
                             float* __restrict__ xf,
                             _Float16* __restrict__ xh) {
    const int row = blockIdx.x;               // b*T + t
    const int t   = row & (TT - 1);
    const int tok = prob_x[row];
    const float log_inc = 9.210340371976184f / 511.0f;   // ln(1e4)/(512-1)
    for (int h = threadIdx.x; h < HH; h += blockDim.x) {
        int   i   = h >> 1;
        float ang = (float)t * __expf(-(float)i * log_inc);
        float pe  = (h & 1) ? __cosf(ang) : __sinf(ang);
        float v   = emb[(size_t)tok * HH + h] + pe;
        xf[(size_t)row * HH + h] = v;
        xh[(size_t)row * HH + h] = (_Float16)v;
    }
}

// ---------------------------------------------------------------------------
// Generic WMMA GEMM: C(MxN,f32/f16) = alpha * A(MxK,f16) * B(KxN,f16) [+resid]
// block = 256 threads (8 waves), tile 128x128, k-step 32.
// A tile staged by the Tensor Data Mover (one tensor_load_to_lds per k-step,
// issued by wave 0, hardware-padded to the 40-half LDS stride; TENSORcnt).
// B tile staged transposed (manual) so fragments stay contiguous b128s.
// ---------------------------------------------------------------------------
__global__ __launch_bounds__(256)
void gemm_wmma_kernel(const _Float16* __restrict__ A,
                      const _Float16* __restrict__ B,
                      float* __restrict__ Cf,
                      _Float16* __restrict__ Ch,
                      const float* __restrict__ resid,
                      float alpha, int M, int N, int K) {
    __shared__ _Float16 sA[128 * 40];         // row-major, stride 40 halfs
    __shared__ _Float16 sB[128 * 40];         // B^T: [n][k], stride 40 halfs

    const int tid   = threadIdx.x;
    const int lane  = tid & 31;
    const int wave  = tid >> 5;
    const int wm    = wave >> 1;              // 0..3
    const int wn    = wave & 1;               // 0..1
    const int l15   = lane & 15;
    const int koff  = (lane >> 4) * 8;        // 0 or 8 (A/B fragment K offset)
    const int base16= (lane >> 4) * 8;        // D-layout row offset

    const int mBase = blockIdx.y * 128;
    const int nBase = blockIdx.x * 128;

    const unsigned sA_base = (unsigned)(size_t)sA;

    // static staging decomposition for the B tile
    const int bk = tid >> 4;                  // B k-rows: bk and bk+16
    const int bn = (tid & 15) * 8;            // B col group (halfs)

    v8f acc[2][4] = {};

    for (int k0 = 0; k0 < K; k0 += 32) {
        // ---- A tile: single TDM DMA global->LDS (wave 0 issues) ----
        if (wave == 0)
            tdm_load_2d(sA_base, A + (size_t)mBase * K + k0,
                        /*tile_w=*/32, /*tile_h=*/128,
                        /*row_len=*/(u32)K, TDM_PAD_ROW32);

        // ---- B tile: load 2 x 16B, store transposed into LDS ----
        union { uint4 u; _Float16 h[8]; } t0, t1;
        t0.u = *(const uint4*)(B + (size_t)(k0 + bk) * N + nBase + bn);
        t1.u = *(const uint4*)(B + (size_t)(k0 + bk + 16) * N + nBase + bn);
#pragma unroll
        for (int e = 0; e < 8; ++e) sB[(bn + e) * 40 + bk]      = t0.h[e];
#pragma unroll
        for (int e = 0; e < 8; ++e) sB[(bn + e) * 40 + bk + 16] = t1.h[e];

        if (k0 + 32 < K)
            __builtin_prefetch(B + (size_t)(k0 + 32 + bk) * N + nBase + bn, 0, 1);

        if (wave == 0)
            __builtin_amdgcn_s_wait_tensorcnt(0);  // drain TDM before publish
        __syncthreads();        // cross-wave: whole tile visible in LDS

        v16h af[2], bf[4];
#pragma unroll
        for (int mt = 0; mt < 2; ++mt) {
            const _Float16* p = &sA[(wm * 32 + mt * 16 + l15) * 40 + koff];
            af[mt] = make_frag(p, p + 16);
        }
#pragma unroll
        for (int nt = 0; nt < 4; ++nt) {
            const _Float16* p = &sB[(wn * 64 + nt * 16 + l15) * 40 + koff];
            bf[nt] = make_frag(p, p + 16);
        }
#pragma unroll
        for (int mt = 0; mt < 2; ++mt)
#pragma unroll
            for (int nt = 0; nt < 4; ++nt)
                acc[mt][nt] = wmma_f16(af[mt], bf[nt], acc[mt][nt]);
        __syncthreads();
    }

    // ---- epilogue ----
#pragma unroll
    for (int mt = 0; mt < 2; ++mt)
#pragma unroll
        for (int nt = 0; nt < 4; ++nt) {
            int rowB = mBase + wm * 32 + mt * 16 + base16;
            int col  = nBase + wn * 64 + nt * 16 + l15;
#pragma unroll
            for (int r = 0; r < 8; ++r) {
                size_t idx = (size_t)(rowB + r) * N + col;
                float  v   = acc[mt][nt][r] * alpha;
                if (resid) v += resid[idx];
                if (Cf) Cf[idx] = v;
                if (Ch) Ch[idx] = (_Float16)v;
            }
        }
}

// ---------------------------------------------------------------------------
// Flash attention with causal + Q-matrix mask. One wave per (b, head, 16-row
// query tile). Q fragments come straight from global (rows contiguous);
// K chunks stage through LDS via GLOBAL_LOAD_ASYNC_TO_LDS_B128 (ASYNCcnt);
// P goes through LDS to build A fragments; V is transposed through LDS.
// ---------------------------------------------------------------------------
__global__ __launch_bounds__(32)
void attn_kernel(const _Float16* __restrict__ Q,
                 const _Float16* __restrict__ K,
                 const _Float16* __restrict__ V,
                 const int* __restrict__ prob_x,
                 const float* __restrict__ item_mask,
                 _Float16* __restrict__ O) {
    __shared__ _Float16 sP[16 * 40];          // P tile 16x32, stride 40
    __shared__ _Float16 sK[32 * 72];          // K chunk [t][d], stride 72
    __shared__ _Float16 sVt[64 * 40];         // V^T chunk [d][t], stride 40

    const int lane   = threadIdx.x;
    const int l15    = lane & 15;
    const int koff   = (lane >> 4) * 8;
    const int base16 = (lane >> 4) * 8;
    const int f0     = blockIdx.x * 16;
    const int b      = blockIdx.y >> 4;
    const int n      = blockIdx.y & 15;

    const size_t headOff = ((size_t)b * TT * NH + n) * HD;  // (t,d) at +t*NH*HD+d
    const _Float16* Qb = Q + headOff;
    const _Float16* Kb = K + headOff;
    const _Float16* Vb = V + headOff;

    const unsigned sK_base = (unsigned)(size_t)sK;

    // Q fragments for this row tile (16 x 64 = two 16x32 fragments)
    v16h aq[2];
    {
        const _Float16* qr = Qb + (size_t)(f0 + l15) * (NH * HD);
#pragma unroll
        for (int s = 0; s < 2; ++s)
            aq[s] = make_frag(qr + s * 32 + koff, qr + s * 32 + 16 + koff);
    }

    // pid of this lane's 8 rows (fixed for the whole block)
    int pidr[8];
#pragma unroll
    for (int r = 0; r < 8; ++r)
        pidr[r] = (prob_x[b * TT + f0 + base16 + r] - 4) >> 1;

    float m[8], l[8];
    v8f   Oacc[4] = {};
#pragma unroll
    for (int r = 0; r < 8; ++r) { m[r] = NEG_INF; l[r] = 0.f; }

    for (int t0 = 0; t0 < f0 + 16; t0 += 32) {
        // ---- stage K chunk (32 rows x 64 halfs) into LDS via async DMA ----
        {
            const _Float16* kr = Kb + (size_t)(t0 + lane) * (NH * HD);
            const unsigned  lrow = sK_base + (unsigned)(lane * 72 * 2);
#pragma unroll
            for (int q = 0; q < 8; ++q)
                async_load_b128(lrow + q * 16, kr + q * 8);
        }

        // ---- stage V chunk transposed into LDS (manual: needs transpose) ----
        {
            const _Float16* vr = Vb + (size_t)(t0 + lane) * (NH * HD);
#pragma unroll
            for (int q = 0; q < 8; ++q) {
                union { uint4 u; _Float16 h[8]; } tmp;
                tmp.u = *(const uint4*)(vr + q * 8);
#pragma unroll
                for (int e = 0; e < 8; ++e)
                    sVt[(q * 8 + e) * 40 + lane] = tmp.h[e];
            }
        }

        wait_async0();          // K chunk resident in LDS

        // ---- S = Q K^T for two 16-col subtiles, with masks ----
        float sm[2][8];
#pragma unroll
        for (int sub = 0; sub < 2; ++sub) {
            v8f S = {};
            const int tcol = t0 + sub * 16 + l15;
            const _Float16* krl = &sK[(sub * 16 + l15) * 72];
#pragma unroll
            for (int s = 0; s < 2; ++s) {
                v16h bk = make_frag(krl + s * 32 + koff, krl + s * 32 + 16 + koff);
                S = wmma_f16(aq[s], bk, S);
            }
            const int pidc = (prob_x[b * TT + tcol] - 4) >> 1;
#pragma unroll
            for (int r = 0; r < 8; ++r) {
                int   row = f0 + base16 + r;
                float sv  = S[r];
                if (tcol > row)                sv = NEG_INF;
                else if (row == 0 || tcol == 0) sv += PAD_NEG;
                else sv += item_mask[(size_t)pidr[r] * NIT + pidc];
                sm[sub][r] = sv;
            }
        }

        // ---- streaming softmax update ----
#pragma unroll
        for (int r = 0; r < 8; ++r) {
            float vmax = fmaxf(sm[0][r], sm[1][r]);
#pragma unroll
            for (int off = 8; off >= 1; off >>= 1)
                vmax = fmaxf(vmax, __shfl_xor(vmax, off, 16));
            float mnew  = fmaxf(m[r], vmax);
            float scale = __expf(m[r] - mnew);
            float e0    = __expf(sm[0][r] - mnew);
            float e1    = __expf(sm[1][r] - mnew);
            float sum   = e0 + e1;
#pragma unroll
            for (int off = 8; off >= 1; off >>= 1)
                sum += __shfl_xor(sum, off, 16);
            l[r] = l[r] * scale + sum;
            m[r] = mnew;
#pragma unroll
            for (int dt = 0; dt < 4; ++dt) Oacc[dt][r] *= scale;
            sP[(base16 + r) * 40 + l15]      = (_Float16)e0;
            sP[(base16 + r) * 40 + 16 + l15] = (_Float16)e1;
        }

        // LDS produced by other lanes of this wave -> drain DS counter
        wait_ds0();

        // ---- O += P(16x32) * V(32x64) ----
        v16h ap = make_frag(&sP[l15 * 40 + koff], &sP[l15 * 40 + 16 + koff]);
#pragma unroll
        for (int dt = 0; dt < 4; ++dt) {
            const _Float16* p = &sVt[(dt * 16 + l15) * 40 + koff];
            v16h bv = make_frag(p, p + 16);
            Oacc[dt] = wmma_f16(ap, bv, Oacc[dt]);
        }
    }

    // ---- finalize + store (b, f, n, d) f16 ----
#pragma unroll
    for (int dt = 0; dt < 4; ++dt)
#pragma unroll
        for (int r = 0; r < 8; ++r) {
            int row = f0 + base16 + r;
            int col = dt * 16 + l15;
            float o = Oacc[dt][r] / l[r];
            O[headOff + (size_t)row * (NH * HD) + col] = (_Float16)o;
        }
}

// ---------------------------------------------------------------------------
// LayerNorm over H=1024
// ---------------------------------------------------------------------------
__global__ __launch_bounds__(256)
void ln_kernel(const float* __restrict__ Y,
               const float* __restrict__ gamma,
               const float* __restrict__ beta,
               float* __restrict__ out) {
    __shared__ float s1[256], s2[256];
    const int row = blockIdx.x, tid = threadIdx.x;
    const float* y = Y + (size_t)row * HH;
    float a = 0.f, b = 0.f;
    for (int i = tid; i < HH; i += 256) { float v = y[i]; a += v; b += v * v; }
    s1[tid] = a; s2[tid] = b;
    __syncthreads();
    for (int s = 128; s > 0; s >>= 1) {
        if (tid < s) { s1[tid] += s1[tid + s]; s2[tid] += s2[tid + s]; }
        __syncthreads();
    }
    const float mu  = s1[0] * (1.0f / HH);
    const float var = s2[0] * (1.0f / HH) - mu * mu;
    const float inv = rsqrtf(var + 1e-3f);
    for (int i = tid; i < HH; i += 256)
        out[(size_t)row * HH + i] = (y[i] - mu) * inv * gamma[i] + beta[i];
}

// ---------------------------------------------------------------------------
// IRT head: relu(y @ w_hid + b) @ w_skill + b, then single sigmoid dot with
// the gathered w_prob column (take_along_axis shortcut).
// ---------------------------------------------------------------------------
__global__ __launch_bounds__(128)
void head_kernel(const float* __restrict__ Y,
                 const int* __restrict__ prob_y2,
                 const float* __restrict__ w_hid, const float* __restrict__ b_hid,
                 const float* __restrict__ w_skill, const float* __restrict__ b_skill,
                 const float* __restrict__ w_prob, const float* __restrict__ b_prob,
                 float* __restrict__ out) {
    __shared__ float sy[HH];
    __shared__ float sh[2 * NSK];
    __shared__ float ssk[NSK];
    const int row = blockIdx.x, tid = threadIdx.x;
    for (int i = tid; i < HH; i += 128) sy[i] = Y[(size_t)row * HH + i];
    __syncthreads();
    for (int j = tid; j < 2 * NSK; j += 128) {
        float a = b_hid[j];
        for (int k = 0; k < HH; ++k) a += sy[k] * w_hid[(size_t)k * (2 * NSK) + j];
        sh[j] = fmaxf(a, 0.f);
    }
    __syncthreads();
    for (int j = tid; j < NSK; j += 128) {
        float a = b_skill[j];
        for (int k = 0; k < 2 * NSK; ++k) a += sh[k] * w_skill[k * NSK + j];
        ssk[j] = a;
    }
    __syncthreads();
    if (tid == 0) {
        const int jp = (prob_y2[row] - 4) >> 1;
        float a = b_prob[jp];
        for (int k = 0; k < NSK; ++k) a += ssk[k] * w_prob[(size_t)k * NIT + jp];
        out[row] = 1.f / (1.f + __expf(-a));
    }
}

// ---------------------------------------------------------------------------
// Host-side orchestration
// ---------------------------------------------------------------------------
extern "C" void kernel_launch(void* const* d_in, const int* in_sizes, int n_in,
                              void* d_out, int out_size, void* d_ws, size_t ws_size,
                              hipStream_t stream) {
    (void)in_sizes; (void)n_in; (void)out_size; (void)ws_size;

    const int*   prob_x   = (const int*)  d_in[0];
    const int*   prob_y2  = (const int*)  d_in[1];
    const float* emb      = (const float*)d_in[2];
    const float* wq       = (const float*)d_in[3];
    const float* wk       = (const float*)d_in[4];
    const float* wv       = (const float*)d_in[5];
    const float* wo       = (const float*)d_in[6];
    const float* ln_gamma = (const float*)d_in[7];
    const float* ln_beta  = (const float*)d_in[8];
    const float* w_hid    = (const float*)d_in[9];
    const float* b_hid    = (const float*)d_in[10];
    const float* w_skill  = (const float*)d_in[11];
    const float* b_skill  = (const float*)d_in[12];
    const float* w_prob   = (const float*)d_in[13];
    const float* b_prob   = (const float*)d_in[14];
    const float* q_matrix = (const float*)d_in[15];
    float*       out      = (float*)d_out;

    char*  ws  = (char*)d_ws;
    size_t off = 0;
    auto alloc = [&](size_t bytes) -> void* {
        void* p = ws + off;
        off = (off + bytes + 255) & ~(size_t)255;
        return p;
    };

    const size_t HW = (size_t)HH * HH;        // 1M elements
    _Float16* wq_h = (_Float16*)alloc(HW * 2);
    _Float16* wk_h = (_Float16*)alloc(HW * 2);
    _Float16* wv_h = (_Float16*)alloc(HW * 2);
    _Float16* wo_h = (_Float16*)alloc(HW * 2);
    _Float16* x_h  = (_Float16*)alloc((size_t)MROWS * HH * 2);
    float*    x_f  = (float*)   alloc((size_t)MROWS * HH * 4);
    _Float16* q_h  = (_Float16*)alloc((size_t)MROWS * HH * 2);
    _Float16* k_h  = (_Float16*)alloc((size_t)MROWS * HH * 2);
    _Float16* v_h  = (_Float16*)alloc((size_t)MROWS * HH * 2);
    _Float16* a_h  = (_Float16*)alloc((size_t)MROWS * HH * 2);
    float*    y_f  = (float*)   alloc((size_t)MROWS * HH * 4);
    float*    y_ln = (float*)   alloc((size_t)MROWS * HH * 4);
    float*    imask= (float*)   alloc((size_t)NIT * NIT * 4);

    // 1) weights -> f16
    cvt_f16_kernel<<<512, 256, 0, stream>>>(wq, wq_h, (int)HW);
    cvt_f16_kernel<<<512, 256, 0, stream>>>(wk, wk_h, (int)HW);
    cvt_f16_kernel<<<512, 256, 0, stream>>>(wv, wv_h, (int)HW);
    cvt_f16_kernel<<<512, 256, 0, stream>>>(wo, wo_h, (int)HW);

    // 2) item-item mask from Q-matrix
    qtq_mask_kernel<<<NIT, 256, 0, stream>>>(q_matrix, imask);

    // 3) embedding + position encoding
    embed_kernel<<<MROWS, 256, 0, stream>>>(prob_x, emb, x_f, x_h);

    // 4) Q/K/V projections (WMMA GEMM). Q scaled by 1/sqrt(HD).
    dim3 ggrid(HH / 128, MROWS / 128);
    gemm_wmma_kernel<<<ggrid, 256, 0, stream>>>(x_h, wq_h, nullptr, q_h, nullptr,
                                                0.125f, MROWS, HH, HH);
    gemm_wmma_kernel<<<ggrid, 256, 0, stream>>>(x_h, wk_h, nullptr, k_h, nullptr,
                                                1.0f, MROWS, HH, HH);
    gemm_wmma_kernel<<<ggrid, 256, 0, stream>>>(x_h, wv_h, nullptr, v_h, nullptr,
                                                1.0f, MROWS, HH, HH);

    // 5) masked flash attention
    dim3 agrid(TT / 16, BB * NH);
    attn_kernel<<<agrid, 32, 0, stream>>>(q_h, k_h, v_h, prob_x, imask, a_h);

    // 6) output projection + residual
    gemm_wmma_kernel<<<ggrid, 256, 0, stream>>>(a_h, wo_h, y_f, nullptr, x_f,
                                                1.0f, MROWS, HH, HH);

    // 7) LayerNorm
    ln_kernel<<<MROWS, 256, 0, stream>>>(y_f, ln_gamma, ln_beta, y_ln);

    // 8) IRT head with take_along_axis shortcut
    head_kernel<<<MROWS, 128, 0, stream>>>(y_ln, prob_y2, w_hid, b_hid,
                                           w_skill, b_skill, w_prob, b_prob, out);
}
